// AutoGNN_51410758533761
// MI455X (gfx1250) — matrix operand order repaired and checked by reading, not compiled, not run on hardware
//
#include <hip/hip_runtime.h>

// ---------------------------------------------------------------------------
// GraphSAGE 2-layer GNN for MI455X (gfx1250, wave32, WMMA)
//   layer: agg = mean(X[neighbors]) ; h = relu(agg @ W^T)   (W: 128x128)
//   out   = h2 @ Wlast^T                                    (Wlast: 40x128)
// GEMMs run as f16 x f16 -> f32 accumulate via v_wmma_f32_16x16x32_f16.
// ---------------------------------------------------------------------------

typedef _Float16 v16h __attribute__((ext_vector_type(16)));
typedef _Float16 v8h  __attribute__((ext_vector_type(8)));
typedef float    v8f  __attribute__((ext_vector_type(8)));

#define LDA 136   // padded LDS row stride in halves (272B -> 4-bank rotate/row)
#define TN   64   // nodes per block
#define FEAT 128
#define SAMP 25
#define NCLS 40

// A fragment (16x32 f16): lane<16 holds row m=lane, K {0..7,16..23};
// lane>=16 holds row m=lane-16, K {8..15,24..31}.  (ISA 7.12.2)
__device__ __forceinline__ v16h frag_A(const _Float16* s, int rowBase, int kb, int lane) {
  const _Float16* p = s + (size_t)(rowBase + (lane & 15)) * LDA + kb * 32 + ((lane >> 4) << 3);
  v8h x = *(const v8h*)p;
  v8h y = *(const v8h*)(p + 16);
  return __builtin_shufflevector(x, y, 0,1,2,3,4,5,6,7,8,9,10,11,12,13,14,15);
}

// B fragment (32x16 f16): lane<16 holds col n=lane, K {0..7,16..23};
// lane>=16 holds col n=lane-16, K {8..15,24..31}.  (matches A fragment)
// Our GEMM is  agg @ W^T, so B column n == row n of W (contiguous in memory).
__device__ __forceinline__ v16h frag_B(const _Float16* s, int rowBase, int kb, int lane) {
  const _Float16* p = s + (size_t)(rowBase + (lane & 15)) * LDA + kb * 32 + ((lane >> 4) << 3);
  v8h x = *(const v8h*)p;
  v8h y = *(const v8h*)(p + 16);
  return __builtin_shufflevector(x, y, 0,1,2,3,4,5,6,7,8,9,10,11,12,13,14,15);
}

// Stage a 128x128 f32 weight matrix into LDS as f16 (padded stride).
__device__ __forceinline__ void load_w128(const float* __restrict__ W, _Float16* sW, int tid) {
#pragma unroll
  for (int k = 0; k < 16; ++k) {                   // 4096 float4 / 256 threads
    int q = tid + k * 256;
    float4 f = ((const float4*)W)[q];
    int r = q >> 5, c = (q & 31) << 2;
    _Float16* p = sW + (size_t)r * LDA + c;
    p[0] = (_Float16)f.x; p[1] = (_Float16)f.y;
    p[2] = (_Float16)f.z; p[3] = (_Float16)f.w;
  }
}

// Gather-mean 64 node rows (128 feats) into LDS f16. 128 lanes cover one
// 512B row exactly -> fully coalesced; rows out of range become zeros.
__device__ __forceinline__ void gather_mean(const float* __restrict__ in,
                                            const int* __restrict__ nbr,
                                            _Float16* sA, int node0, int n, int tid) {
  int col = tid & 127;
  int rh  = tid >> 7;                              // two rows in flight per pass
  for (int p = 0; p < 32; ++p) {
    int row = p * 2 + rh;
    int g   = node0 + row;
    float acc = 0.0f;
    if (g < n) {
      const int* np = nbr + (size_t)g * SAMP;
      int idx[SAMP];
#pragma unroll
      for (int s = 0; s < SAMP; ++s) idx[s] = np[s];
#pragma unroll
      for (int s = 0; s < SAMP; ++s) acc += in[(size_t)idx[s] * FEAT + col];
      acc *= (1.0f / SAMP);
    }
    sA[(size_t)row * LDA + col] = (_Float16)acc;
  }
}

// ---------------- kernel 1: one GNN layer (gather-mean + GEMM + relu) -------
__global__ __launch_bounds__(256) void gnn_layer_kernel(
    const float* __restrict__ in, const int* __restrict__ nbr,
    const float* __restrict__ W, float* __restrict__ out, int n) {
  __shared__ __align__(16) _Float16 sW[128 * LDA];
  __shared__ __align__(16) _Float16 sA[TN * LDA];

  const int tid   = threadIdx.x;
  const int node0 = blockIdx.x * TN;

  load_w128(W, sW, tid);
  gather_mean(in, nbr, sA, node0, n, tid);
  __syncthreads();

  const int lane = tid & 31;
  const int w    = tid >> 5;
  const int rt   = w & 3;            // 16-row tile
  const int ctb  = (w >> 2) << 2;    // first of four 16-col tiles

  v8f c[4] = {};
#pragma unroll
  for (int kb = 0; kb < 4; ++kb) {
    v16h a = frag_A(sA, rt * 16, kb, lane);
#pragma unroll
    for (int i = 0; i < 4; ++i) {
      v16h b = frag_B(sW, (ctb + i) * 16, kb, lane);
      c[i] = __builtin_amdgcn_wmma_f32_16x16x32_f16(false, a, false, b,
                                                    (short)0, c[i], false, false);
    }
  }

  const int m0 = (lane >> 4) << 3;
  const int nn = lane & 15;
#pragma unroll
  for (int i = 0; i < 4; ++i) {
    int ccol = (ctb + i) * 16 + nn;
#pragma unroll
    for (int v = 0; v < 8; ++v) {
      int g = node0 + rt * 16 + v + m0;
      if (g < n) {
        float val = c[i][v];
        out[(size_t)g * FEAT + ccol] = val > 0.0f ? val : 0.0f;
      }
    }
  }
}

// ---------------- kernel 2: layer-2 + classifier, fused (h2 stays in LDS) --
__global__ __launch_bounds__(256) void gnn_fused_kernel(
    const float* __restrict__ h1, const int* __restrict__ nbr,
    const float* __restrict__ W2, const float* __restrict__ Wl,
    float* __restrict__ out, int n) {
  __shared__ __align__(16) _Float16 sW[128 * LDA];   // 34816 B
  __shared__ __align__(16) _Float16 sA[TN * LDA];    // 17408 B
  __shared__ __align__(16) _Float16 sWl[48 * LDA];   // 13056 B (classes padded 40->48)

  const int tid   = threadIdx.x;
  const int node0 = blockIdx.x * TN;

  load_w128(W2, sW, tid);
#pragma unroll
  for (int k = 0; k < 6; ++k) {                      // 48*32 float4 / 256 threads
    int q = tid + k * 256;
    float4 f = make_float4(0.f, 0.f, 0.f, 0.f);
    if (q < NCLS * 32) f = ((const float4*)Wl)[q];
    int r = q >> 5, ccc = (q & 31) << 2;
    _Float16* p = sWl + (size_t)r * LDA + ccc;
    p[0] = (_Float16)f.x; p[1] = (_Float16)f.y;
    p[2] = (_Float16)f.z; p[3] = (_Float16)f.w;
  }
  gather_mean(h1, nbr, sA, node0, n, tid);
  __syncthreads();

  const int lane = tid & 31;
  const int w    = tid >> 5;
  const int rt   = w & 3;
  const int ctb  = (w >> 2) << 2;
  const int m0   = (lane >> 4) << 3;
  const int nn   = lane & 15;

  // ---- layer 2 GEMM + relu -> h2 tiles in registers ----
  v8f c[4] = {};
#pragma unroll
  for (int kb = 0; kb < 4; ++kb) {
    v16h a = frag_A(sA, rt * 16, kb, lane);
#pragma unroll
    for (int i = 0; i < 4; ++i) {
      v16h b = frag_B(sW, (ctb + i) * 16, kb, lane);
      c[i] = __builtin_amdgcn_wmma_f32_16x16x32_f16(false, a, false, b,
                                                    (short)0, c[i], false, false);
    }
  }

  __syncthreads();                                   // everyone done reading sA
  // ---- stage h2 (f16) back into sA for the classifier GEMM ----
#pragma unroll
  for (int i = 0; i < 4; ++i) {
    int ccol = (ctb + i) * 16 + nn;
#pragma unroll
    for (int v = 0; v < 8; ++v) {
      float val = c[i][v];
      sA[(size_t)(rt * 16 + v + m0) * LDA + ccol] = (_Float16)(val > 0.0f ? val : 0.0f);
    }
  }
  __syncthreads();

  // ---- classifier: 4 row-tiles x 3 col-tiles = 12 tiles over 8 waves ----
#pragma unroll
  for (int pass = 0; pass < 2; ++pass) {
    int t = w + pass * 8;
    if (t < 12) {
      int frt = t & 3, fct = t >> 2;
      v8f acc = {};
#pragma unroll
      for (int kb = 0; kb < 4; ++kb) {
        v16h a = frag_A(sA, frt * 16, kb, lane);
        v16h b = frag_B(sWl, fct * 16, kb, lane);
        acc = __builtin_amdgcn_wmma_f32_16x16x32_f16(false, a, false, b,
                                                     (short)0, acc, false, false);
      }
      int ccol = fct * 16 + nn;
#pragma unroll
      for (int v = 0; v < 8; ++v) {
        int g = node0 + frt * 16 + v + m0;
        if (g < n && ccol < NCLS)
          out[(size_t)g * NCLS + ccol] = acc[v];
      }
    }
  }
}

// ---------------------------------------------------------------------------
extern "C" void kernel_launch(void* const* d_in, const int* in_sizes, int n_in,
                              void* d_out, int out_size, void* d_ws, size_t ws_size,
                              hipStream_t stream) {
  const float* X    = (const float*)d_in[0];
  const int*   nbr  = (const int*)d_in[1];
  const float* W1   = (const float*)d_in[2];
  const float* W2   = (const float*)d_in[3];
  const float* Wl   = (const float*)d_in[4];
  float*       out  = (float*)d_out;
  float*       h1   = (float*)d_ws;           // N*128 f32 = 51.2 MB scratch

  const int n    = in_sizes[0] / FEAT;
  const int nblk = (n + TN - 1) / TN;

  gnn_layer_kernel<<<nblk, 256, 0, stream>>>(X, nbr, W1, h1, n);
  gnn_fused_kernel<<<nblk, 256, 0, stream>>>(h1, nbr, W2, Wl, out, n);
}
